// GraphAttentionLayer_62440234549516
// MI455X (gfx1250) — matrix operand order repaired and checked by reading, not compiled, not run on hardware
//
#include <hip/hip_runtime.h>

typedef __attribute__((ext_vector_type(16))) _Float16 v16h;
typedef __attribute__((ext_vector_type(8)))  float    v8f;

#define N_NODES 8192
#define F_IN    512
#define F_OUT   64
#define ALPHA   0.2f

// workspace layout (bytes)
#define WHT_OFF 0u                      // _Float16 WhT[64][8192]  (1 MB)
#define WT_OFF  (1u << 20)              // _Float16 WT [64][512]   (64 KB)
#define F1_OFF  (WT_OFF + 65536u)       // float f1[8192]
#define F2_OFF  (F1_OFF + 32768u)       // float f2[8192]
#define M2_OFF  (F2_OFF + 32768u)       // float M2

static __device__ __forceinline__ float lrelu(float x) {
    return x > 0.f ? x : ALPHA * x;
}

// ---- W[512x64] (row-major) -> WT[64][512] f16 -------------------------------
__global__ void prep_wt(const float* __restrict__ W, _Float16* __restrict__ WT) {
    int idx = blockIdx.x * blockDim.x + threadIdx.x;
    if (idx < F_IN * F_OUT) {
        int k = idx / F_OUT, n = idx % F_OUT;
        WT[n * F_IN + k] = (_Float16)W[idx];
    }
}

// ---- Wh = h @ W via WMMA f16; writes WhT[64][8192] f16 ----------------------
// one wave per 16-row tile; A from h (fp32->f16 on the fly), B from WT.
__global__ void gemm_wh(const float* __restrict__ h,
                        const _Float16* __restrict__ WT,
                        _Float16* __restrict__ WhT) {
    const int lane = threadIdx.x & 31;
    const int wave = threadIdx.x >> 5;
    const int tile = blockIdx.x * (blockDim.x >> 5) + wave;   // 0..511
    const int i0   = tile * 16;
    const int r    = lane & 15;
    const int hi   = lane >> 4;

    v8f c0 = {}, c1 = {}, c2 = {}, c3 = {};
    const float* hrow = h + (size_t)(i0 + r) * F_IN;

    for (int k0 = 0; k0 < F_IN; k0 += 32) {
        // A fragment: row (i0+r); K runs [k0+16hi, +16) contiguous
        union { v16h v; _Float16 e[16]; } A;
        const float* q0 = hrow + k0 + 16 * hi;
        const float* q1 = hrow + k0 + 16 * hi + 8;
#pragma unroll
        for (int t = 0; t < 8; ++t) {
            A.e[t]     = (_Float16)q0[t];
            A.e[8 + t] = (_Float16)q1[t];
        }
        const int kc = k0 + 16 * hi;
        v16h b0 = *(const v16h*)(WT + (size_t)(r)      * F_IN + kc);
        v16h b1 = *(const v16h*)(WT + (size_t)(16 + r) * F_IN + kc);
        v16h b2 = *(const v16h*)(WT + (size_t)(32 + r) * F_IN + kc);
        v16h b3 = *(const v16h*)(WT + (size_t)(48 + r) * F_IN + kc);
        c0 = __builtin_amdgcn_wmma_f32_16x16x32_f16(false, A.v, false, b0, (short)0, c0, false, false);
        c1 = __builtin_amdgcn_wmma_f32_16x16x32_f16(false, A.v, false, b1, (short)0, c1, false, false);
        c2 = __builtin_amdgcn_wmma_f32_16x16x32_f16(false, A.v, false, b2, (short)0, c2, false, false);
        c3 = __builtin_amdgcn_wmma_f32_16x16x32_f16(false, A.v, false, b3, (short)0, c3, false, false);
    }
    // C layout: VGPR rp, lane: row M = rp + 8*hi, col N = r (+16*t per tile)
#pragma unroll
    for (int rp = 0; rp < 8; ++rp) {
        int row = i0 + rp + 8 * hi;
        WhT[(size_t)(r)      * N_NODES + row] = (_Float16)c0[rp];
        WhT[(size_t)(16 + r) * N_NODES + row] = (_Float16)c1[rp];
        WhT[(size_t)(32 + r) * N_NODES + row] = (_Float16)c2[rp];
        WhT[(size_t)(48 + r) * N_NODES + row] = (_Float16)c3[rp];
    }
}

// ---- f1 = Wh . a[:64],  f2 = Wh . a[64:] ------------------------------------
__global__ void attn_prep(const _Float16* __restrict__ WhT,
                          const float* __restrict__ a,
                          float* __restrict__ f1, float* __restrict__ f2) {
    int i = blockIdx.x * blockDim.x + threadIdx.x;
    float s1 = 0.f, s2 = 0.f;
#pragma unroll 8
    for (int n = 0; n < F_OUT; ++n) {
        float w = (float)WhT[(size_t)n * N_NODES + i];
        s1 += w * a[n];
        s2 += w * a[F_OUT + n];
    }
    f1[i] = s1;
    f2[i] = s2;
}

// ---- M2 = max_j f2[j] (single block reduce) ---------------------------------
__global__ void m2_reduce(const float* __restrict__ f2, float* __restrict__ M2) {
    __shared__ float red[256];
    float m = -3.4e38f;
    for (int i = threadIdx.x; i < N_NODES; i += 256) m = fmaxf(m, f2[i]);
    red[threadIdx.x] = m;
    __syncthreads();
    for (int s = 128; s > 0; s >>= 1) {
        if ((int)threadIdx.x < s) red[threadIdx.x] = fmaxf(red[threadIdx.x], red[threadIdx.x + s]);
        __syncthreads();
    }
    if (threadIdx.x == 0) *M2 = red[0];
}

// ---- main fused pass: one adj read, P tile -> WMMA accumulate, normalize ----
// surrogate max m_i = lrelu(f1[i] + M2) >= every row logit (lrelu monotone),
// and softmax is shift-invariant, so no online rescale is needed.
__global__ void gat_main(const int* __restrict__ adj,
                         const _Float16* __restrict__ WhT,
                         const float* __restrict__ f1,
                         const float* __restrict__ f2,
                         const float* __restrict__ M2p,
                         float* __restrict__ out) {
    const int lane = threadIdx.x & 31;
    const int wave = threadIdx.x >> 5;
    const int tile = blockIdx.x * (blockDim.x >> 5) + wave;   // 0..511
    const int i0   = tile * 16;
    const int r    = lane & 15;
    const int hi   = lane >> 4;
    const int row  = i0 + r;          // query row this lane scores

    const float M2 = *M2p;
    const float fi = f1[row];
    const float mi = lrelu(fi + M2);  // per-row safe shift

    const int* adjrow = adj + (size_t)row * N_NODES;

    v8f c0 = {}, c1 = {}, c2 = {}, c3 = {};
    float ssum = 0.f;

    for (int j0 = 0; j0 < N_NODES; j0 += 32) {
        const int jb0 = j0 + 16 * hi;       // K = e + 16*hi       (e = 0..7)
        const int jb1 = j0 + 16 * hi + 8;   // K = e + 16*hi       (e = 8..15)
        union { v16h v; _Float16 e[16]; } A;
#pragma unroll
        for (int t = 0; t < 8; ++t) {
            int j0a = jb0 + t;
            float sc0 = lrelu(fi + f2[j0a]);
            float p0  = (adjrow[j0a] > 0) ? __expf(sc0 - mi) : 0.f;
            ssum += p0;
            A.e[t] = (_Float16)p0;

            int j1a = jb1 + t;
            float sc1 = lrelu(fi + f2[j1a]);
            float p1  = (adjrow[j1a] > 0) ? __expf(sc1 - mi) : 0.f;
            ssum += p1;
            A.e[8 + t] = (_Float16)p1;
        }
        // B tiles: WhT rows are F_out channels -> contiguous 16-f16 per lane
        const int jc = j0 + 16 * hi;
        v16h b0 = *(const v16h*)(WhT + (size_t)(r)      * N_NODES + jc);
        v16h b1 = *(const v16h*)(WhT + (size_t)(16 + r) * N_NODES + jc);
        v16h b2 = *(const v16h*)(WhT + (size_t)(32 + r) * N_NODES + jc);
        v16h b3 = *(const v16h*)(WhT + (size_t)(48 + r) * N_NODES + jc);
        c0 = __builtin_amdgcn_wmma_f32_16x16x32_f16(false, A.v, false, b0, (short)0, c0, false, false);
        c1 = __builtin_amdgcn_wmma_f32_16x16x32_f16(false, A.v, false, b1, (short)0, c1, false, false);
        c2 = __builtin_amdgcn_wmma_f32_16x16x32_f16(false, A.v, false, b2, (short)0, c2, false, false);
        c3 = __builtin_amdgcn_wmma_f32_16x16x32_f16(false, A.v, false, b3, (short)0, c3, false, false);
    }

    // lanes r and r+16 hold disjoint K-partials of row r's exp-sum
    float stot = ssum + __shfl_xor(ssum, 16, 32);

#pragma unroll
    for (int rp = 0; rp < 8; ++rp) {
        float s   = __shfl(stot, rp + 8 * hi, 32);  // lane (rp+8hi) holds row rp+8hi
        float inv = s > 0.f ? 1.0f / s : 0.f;
        float* orow = out + (size_t)(i0 + rp + 8 * hi) * F_OUT;
        orow[r]      = c0[rp] * inv;
        orow[16 + r] = c1[rp] * inv;
        orow[32 + r] = c2[rp] * inv;
        orow[48 + r] = c3[rp] * inv;
    }
}

extern "C" void kernel_launch(void* const* d_in, const int* in_sizes, int n_in,
                              void* d_out, int out_size, void* d_ws, size_t ws_size,
                              hipStream_t stream) {
    const float* h   = (const float*)d_in[0];
    const int*   adj = (const int*)  d_in[1];
    const float* W   = (const float*)d_in[2];
    const float* a   = (const float*)d_in[3];
    float* out = (float*)d_out;

    char* ws = (char*)d_ws;
    _Float16* WhT = (_Float16*)(ws + WHT_OFF);
    _Float16* WT  = (_Float16*)(ws + WT_OFF);
    float* f1 = (float*)(ws + F1_OFF);
    float* f2 = (float*)(ws + F2_OFF);
    float* M2 = (float*)(ws + M2_OFF);

    prep_wt  <<<(F_IN * F_OUT + 255) / 256, 256, 0, stream>>>(W, WT);
    gemm_wh  <<<128, 128, 0, stream>>>(h, WT, WhT);          // 128 blk * 4 waves = 512 tiles
    attn_prep<<<N_NODES / 256, 256, 0, stream>>>(WhT, a, f1, f2);
    m2_reduce<<<1, 256, 0, stream>>>(f2, M2);
    gat_main <<<128, 128, 0, stream>>>(adj, WhT, f1, f2, M2, out);
}
